// YuanSparseMoeBlock_3332894622522
// MI455X (gfx1250) — matrix operand order repaired and checked by reading, not compiled, not run on hardware
//
#include <hip/hip_runtime.h>
#include <hip/hip_bf16.h>

#define TOKENS 2048   // B*S
#define HID    1024
#define FFN    2048
#define NEXP   8
#define BM 64
#define BN 64
#define BK 32

#ifndef __has_builtin
#define __has_builtin(x) 0
#endif
#if __has_builtin(__builtin_amdgcn_global_load_async_to_lds_b128)
#define ASYNC_LDS 1
#else
#define ASYNC_LDS 0
#endif

typedef __attribute__((ext_vector_type(16))) _Float16 v16h;
typedef __attribute__((ext_vector_type(8)))  _Float16 v8h;
typedef __attribute__((ext_vector_type(8)))  float    v8f;

#if ASYNC_LDS
// Matches the builtin's parameter types: int __attribute__((vector_size(16)))
// in the global (AS1) / LDS (AS3) address spaces.
typedef int v4i_async __attribute__((vector_size(16)));
typedef __attribute__((address_space(1))) v4i_async as1_v4i;
typedef __attribute__((address_space(3))) v4i_async as3_v4i;

__device__ __forceinline__ void wait_async0() {
#if __has_builtin(__builtin_amdgcn_s_wait_asynccnt)
  __builtin_amdgcn_s_wait_asynccnt(0);
#else
  asm volatile("s_wait_asynccnt 0" ::: "memory");
#endif
}
#endif

union HV16 { v16h v; v8h h[2]; };

__device__ __forceinline__ v8f wmma_f16(v16h a, v16h b, v8f c) {
  // D = A(16x32 f16) * B(32x16 f16) + C(16x16 f32)
  return __builtin_amdgcn_wmma_f32_16x16x32_f16(false, a, false, b, (short)0, c, false, false);
}

// Stage one 64x32 f16 A-tile row-major into LDS; 16B per thread (256 thr).
// Rows past `rows` are clamped to the last valid row: they only feed C-rows
// that the epilogue masks out, so no zero-fill is required (async-friendly).
__device__ __forceinline__ void stage_a_tile(const _Float16* __restrict__ src_base,
                                             size_t src_stride, int m0, int rows,
                                             int k0, _Float16* lds, int tid) {
  int r = tid >> 2;
  int c = (tid & 3) * 8;
  int gr = m0 + r;
  if (gr >= rows) gr = rows - 1;
  const _Float16* src = src_base + (size_t)gr * src_stride + k0 + c;
#if ASYNC_LDS
  __builtin_amdgcn_global_load_async_to_lds_b128(
      (as1_v4i*)(void*)src, (as3_v4i*)&lds[r * BK + c], 0, 0);
#else
  *(v8h*)&lds[r * BK + c] = *(const v8h*)src;
#endif
}

__device__ __forceinline__ void stage_wait() {
#if ASYNC_LDS
  wait_async0();
#endif
  __syncthreads();
}

// ---------------------------------------------------------------- router ----
__global__ void init_counts(int* cnt) {
  if (threadIdx.x < NEXP) cnt[threadIdx.x] = 0;
}

__global__ void router_kernel(const float* __restrict__ x, const float* __restrict__ rw,
                              int* cnt, int* sel, int* slot, float* wgt) {
  const int wave = threadIdx.x >> 5;
  const int lane = threadIdx.x & 31;
  const int n = blockIdx.x * 8 + wave;     // one wave32 per token
  float acc[NEXP];
#pragma unroll
  for (int e = 0; e < NEXP; ++e) acc[e] = 0.f;
  for (int h = lane; h < HID; h += 32) {
    float xv = x[(size_t)n * HID + h];
    const float4* rp = (const float4*)&rw[(size_t)h * NEXP];
    float4 r0 = rp[0], r1 = rp[1];
    acc[0] += xv * r0.x; acc[1] += xv * r0.y; acc[2] += xv * r0.z; acc[3] += xv * r0.w;
    acc[4] += xv * r1.x; acc[5] += xv * r1.y; acc[6] += xv * r1.z; acc[7] += xv * r1.w;
  }
#pragma unroll
  for (int e = 0; e < NEXP; ++e)
#pragma unroll
    for (int o = 16; o > 0; o >>= 1) acc[e] += __shfl_xor(acc[e], o, 32);

  if (lane == 0) {
    float mx = acc[0];
#pragma unroll
    for (int e = 1; e < NEXP; ++e) mx = fmaxf(mx, acc[e]);
    float pr[NEXP];
#pragma unroll
    for (int e = 0; e < NEXP; ++e) pr[e] = __expf(acc[e] - mx);
    int e0 = 0;
#pragma unroll
    for (int e = 1; e < NEXP; ++e) if (pr[e] > pr[e0]) e0 = e;
    int e1 = (e0 == 0) ? 1 : 0;
#pragma unroll
    for (int e = 0; e < NEXP; ++e) if (e != e0 && pr[e] > pr[e1]) e1 = e;
    float s = pr[e0] + pr[e1];
    int s0 = atomicAdd(&cnt[e0], 1);
    int s1 = atomicAdd(&cnt[e1], 1);
    sel[2*n]   = e0; slot[2*n]   = s0; wgt[2*n]   = pr[e0] / s;
    sel[2*n+1] = e1; slot[2*n+1] = s1; wgt[2*n+1] = pr[e1] / s;
  }
}

__global__ void offsets_kernel(const int* cnt, int* off) {
  if (threadIdx.x == 0) {
    int o = 0;
    for (int e = 0; e < NEXP; ++e) { off[e] = o; o += cnt[e]; }
    off[NEXP] = o;
  }
}

// expert-sorted f16 copy of the activations: Ag[off[e]+slot][:] = x[n][:]
__global__ void gather_kernel(const float* __restrict__ x, const int* __restrict__ sel,
                              const int* __restrict__ slot, const int* __restrict__ off,
                              _Float16* __restrict__ Ag) {
  const int p = blockIdx.x;            // 0 .. 2N-1
  const int n = p >> 1;
  const int r = off[sel[p]] + slot[p];
  const int c = threadIdx.x * 8;       // 128 threads * 8 elems = 1024
  const float4* s = (const float4*)&x[(size_t)n * HID + c];
  float4 f0 = s[0], f1 = s[1];
  v8h hv;
  hv[0] = (_Float16)f0.x; hv[1] = (_Float16)f0.y; hv[2] = (_Float16)f0.z; hv[3] = (_Float16)f0.w;
  hv[4] = (_Float16)f1.x; hv[5] = (_Float16)f1.y; hv[6] = (_Float16)f1.z; hv[7] = (_Float16)f1.w;
  *(v8h*)&Ag[(size_t)r * HID + c] = hv;
}

// ------------------------------------------------------------------ GEMM1 ---
// For expert e: C = Ag_rows x W1[e] (gate cols n0..n0+63 AND value cols n0+F..)
// fused SwiGLU -> G (f16). 256 thr = 8 waves; wave (wm = w&3, wn = w>>2) owns a
// 16-row strip and two 16-col subtiles of each half.
__global__ void gemm1_kernel(const _Float16* __restrict__ Ag, const float* __restrict__ W1,
                             const int* __restrict__ cnt, const int* __restrict__ off,
                             _Float16* __restrict__ G) {
  const int e    = blockIdx.z;
  const int m0   = blockIdx.x * BM;
  const int n0   = blockIdx.y * BN;          // gate column base in [0, FFN)
  const int rows = cnt[e];
  if (m0 >= rows) return;
  const int rbase = off[e];
  const float* W1e = W1 + (size_t)e * HID * (2 * FFN);

  __shared__ __align__(32) _Float16 Al[BM * BK];
  __shared__ __align__(32) _Float16 Bg[BN * BK];   // transposed: [col][k]
  __shared__ __align__(32) _Float16 Bv[BN * BK];

  const int tid  = threadIdx.x;
  const int lane = tid & 31, wave = tid >> 5;
  const int wm = wave & 3, wn = wave >> 2;
  const int hsel = lane >> 4, lc = lane & 15;

  v8f ag0 = {}, ag1 = {}, av0 = {}, av1 = {};

  for (int k0 = 0; k0 < HID; k0 += BK) {
    // A tile (f16, pure copy -> async global->LDS when available)
    stage_a_tile(Ag + (size_t)rbase * HID, HID, m0, rows, k0, Al, tid);
    // stage gate + value weight tiles (32x64 fp32 each), convert+transpose to f16 LDS
#pragma unroll
    for (int it = 0; it < 2; ++it) {
      int idx = tid + it * 256;          // 0..511
      int kk  = idx >> 4;                // 0..31
      int c4  = (idx & 15) * 4;          // 0..60
      const float* bp = &W1e[(size_t)(k0 + kk) * (2 * FFN) + n0 + c4];
      float4 g4 = *(const float4*)bp;
      float4 v4 = *(const float4*)(bp + FFN);
      Bg[(c4 + 0) * BK + kk] = (_Float16)g4.x;
      Bg[(c4 + 1) * BK + kk] = (_Float16)g4.y;
      Bg[(c4 + 2) * BK + kk] = (_Float16)g4.z;
      Bg[(c4 + 3) * BK + kk] = (_Float16)g4.w;
      Bv[(c4 + 0) * BK + kk] = (_Float16)v4.x;
      Bv[(c4 + 1) * BK + kk] = (_Float16)v4.y;
      Bv[(c4 + 2) * BK + kk] = (_Float16)v4.z;
      Bv[(c4 + 3) * BK + kk] = (_Float16)v4.w;
    }
    if (k0 + BK < HID && tid < BK)  // hint next weight rows
      __builtin_prefetch(&W1e[(size_t)(k0 + BK + tid) * (2 * FFN) + n0], 0, 1);
    stage_wait();

    HV16 a, bg, bv;
    const int mrow = wm * 16 + lc;
    a.h[0] = *(const v8h*)&Al[mrow * BK + hsel * 8];        // K = hsel*8 .. +7
    a.h[1] = *(const v8h*)&Al[mrow * BK + 16 + hsel * 8];   // K = 16+hsel*8 .. +7
#pragma unroll
    for (int t = 0; t < 2; ++t) {
      const int col = (wn * 32 + t * 16 + lc) * BK + hsel * 8;
      bg.h[0] = *(const v8h*)&Bg[col];
      bg.h[1] = *(const v8h*)&Bg[col + 16];
      bv.h[0] = *(const v8h*)&Bv[col];
      bv.h[1] = *(const v8h*)&Bv[col + 16];
      if (t == 0) { ag0 = wmma_f16(a.v, bg.v, ag0); av0 = wmma_f16(a.v, bv.v, av0); }
      else        { ag1 = wmma_f16(a.v, bg.v, ag1); av1 = wmma_f16(a.v, bv.v, av1); }
    }
    __syncthreads();
  }
  // fused SwiGLU epilogue: g = silu(gate) * value -> G (f16)
#pragma unroll
  for (int t = 0; t < 2; ++t) {
    v8f agv = t ? ag1 : ag0;
    v8f avv = t ? av1 : av0;
    const int gcol = n0 + wn * 32 + t * 16 + lc;
#pragma unroll
    for (int r = 0; r < 8; ++r) {
      int grow = m0 + wm * 16 + hsel * 8 + r;
      if (grow < rows) {
        float ga = agv[r];
        float g  = (ga / (1.f + __expf(-ga))) * avv[r];
        G[(size_t)(rbase + grow) * FFN + gcol] = (_Float16)g;
      }
    }
  }
}

// ------------------------------------------------------------------ GEMM2 ---
// Y2 = G_rows x W2[e]  (fp32 out, combined later)
__global__ void gemm2_kernel(const _Float16* __restrict__ G, const float* __restrict__ W2,
                             const int* __restrict__ cnt, const int* __restrict__ off,
                             float* __restrict__ Y2) {
  const int e    = blockIdx.z;
  const int m0   = blockIdx.x * BM;
  const int n0   = blockIdx.y * BN;          // output column base in [0, HID)
  const int rows = cnt[e];
  if (m0 >= rows) return;
  const int rbase = off[e];
  const float* W2e = W2 + (size_t)e * FFN * HID;

  __shared__ __align__(32) _Float16 Al[BM * BK];
  __shared__ __align__(32) _Float16 Bt[BN * BK];

  const int tid  = threadIdx.x;
  const int lane = tid & 31, wave = tid >> 5;
  const int wm = wave & 3, wn = wave >> 2;
  const int hsel = lane >> 4, lc = lane & 15;

  v8f c0 = {}, c1 = {};

  for (int k0 = 0; k0 < FFN; k0 += BK) {
    stage_a_tile(G + (size_t)rbase * FFN, FFN, m0, rows, k0, Al, tid);
#pragma unroll
    for (int it = 0; it < 2; ++it) {
      int idx = tid + it * 256;
      int kk  = idx >> 4;
      int c4  = (idx & 15) * 4;
      float4 b4 = *(const float4*)&W2e[(size_t)(k0 + kk) * HID + n0 + c4];
      Bt[(c4 + 0) * BK + kk] = (_Float16)b4.x;
      Bt[(c4 + 1) * BK + kk] = (_Float16)b4.y;
      Bt[(c4 + 2) * BK + kk] = (_Float16)b4.z;
      Bt[(c4 + 3) * BK + kk] = (_Float16)b4.w;
    }
    if (k0 + BK < FFN && tid < BK)
      __builtin_prefetch(&W2e[(size_t)(k0 + BK + tid) * HID + n0], 0, 1);
    stage_wait();

    HV16 a, b;
    const int mrow = wm * 16 + lc;
    a.h[0] = *(const v8h*)&Al[mrow * BK + hsel * 8];
    a.h[1] = *(const v8h*)&Al[mrow * BK + 16 + hsel * 8];
#pragma unroll
    for (int t = 0; t < 2; ++t) {
      const int col = (wn * 32 + t * 16 + lc) * BK + hsel * 8;
      b.h[0] = *(const v8h*)&Bt[col];
      b.h[1] = *(const v8h*)&Bt[col + 16];
      if (t == 0) c0 = wmma_f16(a.v, b.v, c0);
      else        c1 = wmma_f16(a.v, b.v, c1);
    }
    __syncthreads();
  }
#pragma unroll
  for (int t = 0; t < 2; ++t) {
    v8f cv = t ? c1 : c0;
    const int gcol = n0 + wn * 32 + t * 16 + lc;
#pragma unroll
    for (int r = 0; r < 8; ++r) {
      int grow = m0 + wm * 16 + hsel * 8 + r;
      if (grow < rows)
        Y2[(size_t)(rbase + grow) * HID + gcol] = cv[r];
    }
  }
}

// --------------------------------------------------------------- combine ----
__global__ void combine_kernel(const float* __restrict__ Y2, const int* __restrict__ sel,
                               const int* __restrict__ slot, const int* __restrict__ off,
                               const float* __restrict__ wgt, float* __restrict__ out) {
  const int n  = blockIdx.x;
  const int r0 = off[sel[2*n]]   + slot[2*n];
  const int r1 = off[sel[2*n+1]] + slot[2*n+1];
  const float w0 = wgt[2*n], w1 = wgt[2*n+1];
  const int c = threadIdx.x * 4;
  float4 y0 = *(const float4*)&Y2[(size_t)r0 * HID + c];
  float4 y1 = *(const float4*)&Y2[(size_t)r1 * HID + c];
  float4 o;
  o.x = w0 * y0.x + w1 * y1.x;
  o.y = w0 * y0.y + w1 * y1.y;
  o.z = w0 * y0.z + w1 * y1.z;
  o.w = w0 * y0.w + w1 * y1.w;
  *(float4*)&out[(size_t)n * HID + c] = o;
}

// ------------------------------------------------------------------ launch --
extern "C" void kernel_launch(void* const* d_in, const int* in_sizes, int n_in,
                              void* d_out, int out_size, void* d_ws, size_t ws_size,
                              hipStream_t stream) {
  const float* x  = (const float*)d_in[0];   // [N, H] fp32
  const float* rw = (const float*)d_in[1];   // [H, E]
  const float* W1 = (const float*)d_in[2];   // [E, H, 2F]
  const float* W2 = (const float*)d_in[3];   // [E, F, H]
  float* out = (float*)d_out;

  char* ws = (char*)d_ws;
  size_t o = 0;
  auto take = [&](size_t bytes) { size_t a = o; o += (bytes + 255) & ~(size_t)255; return a; };
  int*      cnt  = (int*)     (ws + take(NEXP * 4));
  int*      off  = (int*)     (ws + take((NEXP + 1) * 4));
  int*      sel  = (int*)     (ws + take((size_t)2 * TOKENS * 4));
  int*      slot = (int*)     (ws + take((size_t)2 * TOKENS * 4));
  float*    wgt  = (float*)   (ws + take((size_t)2 * TOKENS * 4));
  _Float16* Ag   = (_Float16*)(ws + take((size_t)2 * TOKENS * HID * 2));  // 8 MB
  _Float16* G    = (_Float16*)(ws + take((size_t)2 * TOKENS * FFN * 2));  // 16 MB
  float*    Y2   = (float*)   (ws + take((size_t)2 * TOKENS * HID * 4));  // 16 MB
  (void)ws_size; (void)in_sizes; (void)n_in; (void)out_size;

  init_counts<<<1, 32, 0, stream>>>(cnt);
  router_kernel<<<TOKENS / 8, 256, 0, stream>>>(x, rw, cnt, sel, slot, wgt);
  offsets_kernel<<<1, 32, 0, stream>>>(cnt, off);
  gather_kernel<<<2 * TOKENS, HID / 8, 0, stream>>>(x, sel, slot, off, Ag);
  gemm1_kernel<<<dim3(TOKENS / BM, FFN / BN, NEXP), 256, 0, stream>>>(Ag, W1, cnt, off, G);
  gemm2_kernel<<<dim3(TOKENS / BM, HID / BN, NEXP), 256, 0, stream>>>(G, W2, cnt, off, Y2);
  combine_kernel<<<TOKENS, 256, 0, stream>>>(Y2, sel, slot, off, wgt, out);
}
